// TiConv2d_11785390260260
// MI455X (gfx1250) — 8-bit matrix operand order repaired and checked by reading, compiled, not run on hardware
//
#include <hip/hip_runtime.h>
#include <stdint.h>


typedef int v8i __attribute__((ext_vector_type(8)));

#define CI    32
#define CO    32
#define HH    128
#define WW    128
#define KK    288
#define NSTEP 5
#define NCOL  132
#define RSTRIDE (NCOL * CI)

__global__ __launch_bounds__(256)
void conv_wmma_iu8(const int* __restrict__ act, const int* __restrict__ wgt,
                   int* __restrict__ accOut, unsigned* __restrict__ gmax)
{
    __shared__ __align__(16) signed char actLDS[3 * RSTRIDE];
    __shared__ __align__(16) int aImg[NSTEP][2][32][8];
    __shared__ unsigned redMax;

    const int tid = threadIdx.x;
    const int n   = blockIdx.x >> 7;
    const int oh  = blockIdx.x & 127;

    if (tid == 0) redMax = 0u;

    for (int slot = tid; slot < CI * 3 * 130; slot += 256) {
        int ci  = slot / 390;
        int rem = slot - ci * 390;
        int r   = rem / 130;
        int col = rem - r * 130;
        int ih  = oh - 1 + r;
        int iw  = col - 1;
        signed char v = 0;
        if (ih >= 0 && ih < HH && iw >= 0 && iw < WW)
            v = (signed char)act[(((n * CI) + ci) * HH + ih) * WW + iw];
        actLDS[r * RSTRIDE + col * CI + ci] = v;
    }

    for (int idx = tid; idx < NSTEP * 2 * 32 * 8; idx += 256) {
        int s    = idx / 640;
        int r0   = idx - s * 640;
        int mt   = r0 / 320;
        int r1   = r0 - mt * 320;
        int lane = r1 >> 3;
        int d    = r1 & 7;
        int m    = mt * 16 + (lane & 15);
        int kb   = ((d & 3) << 2) + ((d >> 2) << 5)
                 + ((lane < 16) ? 0 : 16) + s * 64;
        unsigned pack = 0u;
        #pragma unroll
        for (int b = 0; b < 4; ++b) {
            int kp = kb + b;
            unsigned byte = 0u;
            if (kp < KK) {
                int ci = kp & 31;
                int g  = kp >> 5;
                int kh = g / 3;
                int kw = g - 3 * kh;
                byte = (unsigned)wgt[m * KK + ci * 9 + kh * 3 + kw] & 0xFFu;
            }
            pack |= byte << (8 * b);
        }
        aImg[s][mt][lane][d] = (int)pack;
    }
    __syncthreads();

    const int wave    = tid >> 5;
    const int lane    = tid & 31;
    const int pix     = (wave << 4) + (lane & 15);
    const int halfOff = (lane >= 16) ? 16 : 0;

    v8i acc0 = {0, 0, 0, 0, 0, 0, 0, 0};
    v8i acc1 = {0, 0, 0, 0, 0, 0, 0, 0};

    #pragma unroll
    for (int s = 0; s < NSTEP; ++s) {
        v8i bfrag;
        {
            constexpr int gTbl[10][2] = {{0,0},{0,1},{0,2},{1,0},{1,1},
                                         {1,2},{2,0},{2,1},{2,2},{0,0}};
            const int g0 = 2 * s;
            const int a0 = gTbl[g0][0] * RSTRIDE + (pix + gTbl[g0][1]) * CI + halfOff;
            const int4 lo = *(const int4*)(actLDS + a0);
            bfrag[0] = lo.x; bfrag[1] = lo.y; bfrag[2] = lo.z; bfrag[3] = lo.w;
            if (2 * s + 1 < 9) {
                const int g1 = 2 * s + 1;
                const int a1 = gTbl[g1][0] * RSTRIDE + (pix + gTbl[g1][1]) * CI + halfOff;
                const int4 hi = *(const int4*)(actLDS + a1);
                bfrag[4] = hi.x; bfrag[5] = hi.y; bfrag[6] = hi.z; bfrag[7] = hi.w;
            } else {
                bfrag[4] = 0; bfrag[5] = 0; bfrag[6] = 0; bfrag[7] = 0;
            }
        }
        v8i a0f, a1f;
        #pragma unroll
        for (int j = 0; j < 8; ++j) {
            a0f[j] = aImg[s][0][lane][j];
            a1f[j] = aImg[s][1][lane][j];
        }
        acc0 = __builtin_amdgcn_wmma_i32_16x16x64_iu8(true, a0f, true, bfrag, acc0, false, false);
        acc1 = __builtin_amdgcn_wmma_i32_16x16x64_iu8(true, a1f, true, bfrag, acc1, false, false);
    }

    unsigned lmax = 0u;
    const int rowAdd = (lane >= 16) ? 8 : 0;
    #pragma unroll
    for (int j = 0; j < 8; ++j) {
        int co0 = j + rowAdd;
        int co1 = 16 + j + rowAdd;
        int v0 = acc0[j], v1 = acc1[j];
        accOut[(((n * CO) + co0) * HH + oh) * WW + pix] = v0;
        accOut[(((n * CO) + co1) * HH + oh) * WW + pix] = v1;
        unsigned u0 = (unsigned)(v0 < 0 ? -v0 : v0);
        unsigned u1 = (unsigned)(v1 < 0 ? -v1 : v1);
        unsigned u  = u0 > u1 ? u0 : u1;
        if (u > lmax) lmax = u;
    }
    atomicMax(&redMax, lmax);
    __syncthreads();
    if (tid == 0) atomicMax(gmax, redMax);
}

__global__ void init_max(unsigned* gmax) { *gmax = 0u; }

__global__ __launch_bounds__(256)
void requant(const unsigned* __restrict__ gmax, const int* __restrict__ expin,
             const int* __restrict__ wexp, float* __restrict__ out, int total)
{
    const unsigned r = *gmax;
    const int bw    = (r == 0u) ? 0 : (32 - __clz((int)(r - 1u)));
    const int shift = bw - 7;
    const int eff   = (shift > 1) ? shift : 2;

    const int idx4 = (blockIdx.x * 256 + threadIdx.x) * 4;
    const int4 a = *(const int4*)((const int*)out + idx4);
    int vals[4] = {a.x, a.y, a.z, a.w};
    float res[4];
    #pragma unroll
    for (int i = 0; i < 4; ++i) {
        int x = vals[i];
        signed char q;
        if (shift >= 1) {
            int rt   = x >> eff;
            int prob = x - (rt << eff);
            int half = 1 << (eff >> 1);
            int qp   = prob / half;
            int pr   = prob - qp * half;
            if (eff & 1) pr <<= 1;
            int sgn  = (x > 0) - (x < 0);
            int y    = rt + ((qp <= pr) ? 0 : sgn);
            y = y < -127 ? -127 : (y > 127 ? 127 : y);
            q = (signed char)y;
        } else {
            q = (signed char)x;
        }
        res[i] = (float)q;
    }
    float4 o; o.x = res[0]; o.y = res[1]; o.z = res[2]; o.w = res[3];
    *(float4*)(out + idx4) = o;

    if (blockIdx.x == 0 && threadIdx.x == 0) {
        int add = (shift > 1) ? shift : ((shift == 1) ? 2 : 0);
        signed char e = (signed char)(expin[0] + wexp[0] + add);
        out[total] = (float)e;
    }
}

extern "C" void kernel_launch(void* const* d_in, const int* in_sizes, int n_in,
                              void* d_out, int out_size, void* d_ws, size_t ws_size,
                              hipStream_t stream)
{
    (void)in_sizes; (void)n_in; (void)out_size; (void)ws_size;
    const int* act  = (const int*)d_in[0];
    const int* expi = (const int*)d_in[1];
    const int* wgt  = (const int*)d_in[2];
    const int* wexp = (const int*)d_in[3];

    unsigned* gmax = (unsigned*)d_ws;
    const int total = 32 * 32 * 128 * 128;

    init_max<<<1, 1, 0, stream>>>(gmax);
    conv_wmma_iu8<<<32 * 128, 256, 0, stream>>>(act, wgt, (int*)d_out, gmax);
    requant<<<total / (256 * 4), 256, 0, stream>>>(gmax, expi, wexp, (float*)d_out, total);
}
